// PerceiverAttention_17970143167102
// MI455X (gfx1250) — matrix operand order repaired and checked by reading, not compiled, not run on hardware
//
#include <hip/hip_runtime.h>
#include <hip/hip_bf16.h>
#include <stdint.h>

// ---------------------------------------------------------------------------
// Types for CDNA5 WMMA (wave32) and TDM descriptors
// ---------------------------------------------------------------------------
typedef __bf16  v4bf  __attribute__((ext_vector_type(4)));
typedef __bf16  v8bf  __attribute__((ext_vector_type(8)));
typedef __bf16  v16bf __attribute__((ext_vector_type(16)));
typedef float   v8f   __attribute__((ext_vector_type(8)));
typedef unsigned int tdm_u32x4 __attribute__((ext_vector_type(4)));
typedef int          tdm_i32x8 __attribute__((ext_vector_type(8)));
typedef int          tdm_i32x4 __attribute__((ext_vector_type(4)));

#if defined(__has_builtin)
#if __has_builtin(__builtin_amdgcn_tensor_load_to_lds) && \
    __has_builtin(__builtin_amdgcn_s_wait_tensorcnt)
#define HAVE_TDM 1
#endif
#endif
#ifndef HAVE_TDM
#define HAVE_TDM 0
#endif

#define B_BATCH 8
#define F_FEAT  8192
#define N_LAT   64
#define D_MODEL 1024
#define INNER   512
#define HEADS   8
#define DHEAD   64
#define KVLEN   8256          // F_FEAT + N_LAT
#define SCALE_Q 0.125f        // 64^-0.5
#define NEG_BIG (-1.0e30f)

__device__ __forceinline__ v8f vzero8() {
  v8f z = {0.f,0.f,0.f,0.f,0.f,0.f,0.f,0.f};
  return z;
}

// D = A(16x32 bf16) * B(32x16 bf16) + C(16x16 f32)
__device__ __forceinline__ v8f wmma_bf16(v16bf a, v16bf b, v8f c) {
  return __builtin_amdgcn_wmma_f32_16x16x32_bf16(
      /*neg_a=*/false, a, /*neg_b=*/false, b,
      /*c_mod=*/(short)0, c, /*reuse_a=*/false, /*reuse_b=*/false);
}

// A-fragment: lane holds row M = (lane&15). j=0..7 -> K = half*8 + j,
// j=8..15 -> K = half*8 + 16 + (j-8).  p points at row + kc + half*8.
__device__ __forceinline__ v16bf load_frag_a(const __bf16* p) {
  v8bf lo = *(const v8bf*)(p);
  v8bf hi = *(const v8bf*)(p + 16);
  return __builtin_shufflevector(lo, hi, 0,1,2,3,4,5,6,7,8,9,10,11,12,13,14,15);
}

// B-fragment: lane holds column N = (lane&15). j=0..7 -> K = half*8 + j,
// j=8..15 -> K = half*8 + 16 + (j-8).  p points at col + kc + half*8.
__device__ __forceinline__ v16bf load_frag_b(const __bf16* p) {
  v8bf lo = *(const v8bf*)(p);
  v8bf hi = *(const v8bf*)(p + 16);
  return __builtin_shufflevector(lo, hi, 0,1,2,3,4,5,6,7,8,9,10,11,12,13,14,15);
}

// ---------------------------------------------------------------------------
// TDM: DMA a 2-D bf16 tile (128 rows x 32 elems, row stride = row_len elems)
// from global into LDS (packed 128x32).  CDNA5 ISA layout.
// Toolchain uses the 6-arg builtin: (g0 u32x4, g1 i32x8, g2 i32x4, g3 i32x4,
// extra i32x8, cpol i32).
// ---------------------------------------------------------------------------
#if HAVE_TDM
__device__ __forceinline__ void tdm_load_tile_2d(const __bf16* gsrc,
                                                 unsigned int lds_off,
                                                 unsigned int row_len_elems) {
  unsigned long long ga = (unsigned long long)(uintptr_t)gsrc;
  tdm_u32x4 g0;
  g0[0] = 1u;                                  // count=1, user descriptor
  g0[1] = lds_off;                             // lds_addr (bytes)
  g0[2] = (unsigned int)ga;                    // global_addr[31:0]
  g0[3] = (unsigned int)((ga >> 32) & 0x01FFFFFFu) | (2u << 30); // addr|type=2
  const unsigned int t_dim0 = 32u;             // tensor dim0 == tile dim0
  const unsigned int t_dim1 = 128u;            // tensor dim1 == tile dim1
  const unsigned int tile0 = 32u, tile1 = 128u;
  unsigned long long stride0 = (unsigned long long)row_len_elems; // 48-bit
  tdm_i32x8 g1;
  g1[0] = (int)(1u << 16);                     // wg_mask=0, data_size=1 (2B)
  g1[1] = (int)((t_dim0 & 0xFFFFu) << 16);     // atomic_addr=0 | dim0[15:0]
  g1[2] = (int)((t_dim0 >> 16) | ((t_dim1 & 0xFFFFu) << 16));
  g1[3] = (int)(((t_dim1 >> 16) & 0xFFFFu) | (tile0 << 16));
  g1[4] = (int)(tile1 & 0xFFFFu);              // tile_dim1 | tile_dim2=0
  g1[5] = (int)(unsigned int)stride0;          // dim0_stride[31:0]
  g1[6] = (int)((unsigned int)(stride0 >> 32) & 0xFFFFu); // hi | dim1_stride lo=0
  g1[7] = 0;                                   // dim1_stride hi
  tdm_i32x4 z4 = {0, 0, 0, 0};                 // groups 2/3 unused (2-D)
  tdm_i32x8 z8 = {0, 0, 0, 0, 0, 0, 0, 0};
  __builtin_amdgcn_tensor_load_to_lds(g0, g1, z4, z4, z8, 0);
}
#endif

// ---------------------------------------------------------------------------
// Kernel 1: f32 -> bf16 convert + transpose  (dst[n][k] = src[k][n])
// ---------------------------------------------------------------------------
__global__ __launch_bounds__(256) void cvt_transpose_kernel(
    const float* __restrict__ src, __bf16* __restrict__ dst, int K, int N) {
  size_t idx = (size_t)blockIdx.x * 256 + threadIdx.x;
  if (idx >= (size_t)K * N) return;
  int k = (int)(idx / N);
  int n = (int)(idx % N);
  dst[(size_t)n * K + k] = (__bf16)src[idx];
}

// ---------------------------------------------------------------------------
// Kernel 2: LayerNorm of x -> bf16 kv rows [0, F_FEAT)
// ---------------------------------------------------------------------------
__global__ __launch_bounds__(256) void ln_media_kernel(
    const float* __restrict__ x, const float* __restrict__ gamma,
    const float* __restrict__ beta, __bf16* __restrict__ kv) {
  int row = blockIdx.x;                 // 0 .. B*F-1
  int b = row / F_FEAT, f = row % F_FEAT;
  int t = threadIdx.x;
  const float4* xr = (const float4*)(x + (size_t)row * D_MODEL);
  float4 v = xr[t];
  float s  = v.x + v.y + v.z + v.w;
  float s2 = v.x*v.x + v.y*v.y + v.z*v.z + v.w*v.w;
  __shared__ float red[512];
  red[t] = s; red[256 + t] = s2;
  __syncthreads();
  for (int o = 128; o > 0; o >>= 1) {
    if (t < o) { red[t] += red[t + o]; red[256 + t] += red[256 + t + o]; }
    __syncthreads();
  }
  float mu  = red[0] * (1.0f / D_MODEL);
  float var = red[256] * (1.0f / D_MODEL) - mu * mu;
  float rstd = rsqrtf(var + 1e-5f);
  int c = t * 4;
  v4bf o4;
  o4[0] = (__bf16)((v.x - mu) * rstd * gamma[c + 0] + beta[c + 0]);
  o4[1] = (__bf16)((v.y - mu) * rstd * gamma[c + 1] + beta[c + 1]);
  o4[2] = (__bf16)((v.z - mu) * rstd * gamma[c + 2] + beta[c + 2]);
  o4[3] = (__bf16)((v.w - mu) * rstd * gamma[c + 3] + beta[c + 3]);
  *(v4bf*)(kv + ((size_t)b * KVLEN + f) * D_MODEL + c) = o4;
}

// ---------------------------------------------------------------------------
// Kernel 3: LayerNorm of latents -> bf16 lat_ln (for Q) and gated kv tail
// ---------------------------------------------------------------------------
__global__ __launch_bounds__(256) void ln_latents_kernel(
    const float* __restrict__ lat, const float* __restrict__ gamma,
    const float* __restrict__ beta, const float* __restrict__ kv_gate,
    __bf16* __restrict__ lat_ln, __bf16* __restrict__ kv) {
  int row = blockIdx.x;                 // 0 .. B*N-1
  int b = row / N_LAT, i = row % N_LAT;
  int t = threadIdx.x;
  float gate = kv_gate[0];
  const float4* xr = (const float4*)(lat + (size_t)row * D_MODEL);
  float4 v = xr[t];
  float s  = v.x + v.y + v.z + v.w;
  float s2 = v.x*v.x + v.y*v.y + v.z*v.z + v.w*v.w;
  __shared__ float red[512];
  red[t] = s; red[256 + t] = s2;
  __syncthreads();
  for (int o = 128; o > 0; o >>= 1) {
    if (t < o) { red[t] += red[t + o]; red[256 + t] += red[256 + t + o]; }
    __syncthreads();
  }
  float mu  = red[0] * (1.0f / D_MODEL);
  float var = red[256] * (1.0f / D_MODEL) - mu * mu;
  float rstd = rsqrtf(var + 1e-5f);
  int c = t * 4;
  float y0 = (v.x - mu) * rstd * gamma[c + 0] + beta[c + 0];
  float y1 = (v.y - mu) * rstd * gamma[c + 1] + beta[c + 1];
  float y2 = (v.z - mu) * rstd * gamma[c + 2] + beta[c + 2];
  float y3 = (v.w - mu) * rstd * gamma[c + 3] + beta[c + 3];
  v4bf a; a[0]=(__bf16)y0; a[1]=(__bf16)y1; a[2]=(__bf16)y2; a[3]=(__bf16)y3;
  *(v4bf*)(lat_ln + (size_t)row * D_MODEL + c) = a;
  v4bf g; g[0]=(__bf16)(y0*gate); g[1]=(__bf16)(y1*gate);
          g[2]=(__bf16)(y2*gate); g[3]=(__bf16)(y3*gate);
  *(v4bf*)(kv + ((size_t)b * KVLEN + F_FEAT + i) * D_MODEL + c) = g;
}

// ---------------------------------------------------------------------------
// Kernel 4: bf16 GEMM  C(MxN) = A(MxK) * BT(NxK)^T, 128x128 block, 4 waves.
// TDM double-buffers 128x32 A- and B-tiles into LDS (8 KB each, 2 buffers);
// wave 0 issues tensor_load_to_lds + s_wait_tensorcnt, all waves barrier.
// ---------------------------------------------------------------------------
template <bool WRITE_F32, bool TRANS_C>
__global__ __launch_bounds__(128) void gemm_bf16_kernel(
    const __bf16* __restrict__ A, const __bf16* __restrict__ BT,
    void* __restrict__ C, int M, int N, int K, float scale) {
  int wave = threadIdx.x >> 5;
  int lane = threadIdx.x & 31;
  int half = lane >> 4;
  int lr   = lane & 15;
  int rblk = blockIdx.x * 128;
  int cblk = blockIdx.y * 128;
  int rbase = rblk + (wave >> 1) * 64;
  int cbase = cblk + (wave & 1) * 64;

  v8f acc[4][4];
#pragma unroll
  for (int mt = 0; mt < 4; mt++)
#pragma unroll
    for (int nt = 0; nt < 4; nt++) acc[mt][nt] = vzero8();

#if HAVE_TDM
  // packed tiles: [row 0..127][k 0..31], row stride 32 elems (64 B)
  __shared__ __align__(16) __bf16 ldsA[2][128 * 32];
  __shared__ __align__(16) __bf16 ldsB[2][128 * 32];
  unsigned int offA[2], offB[2];
  offA[0] = (unsigned int)(uintptr_t)&ldsA[0][0];
  offA[1] = (unsigned int)(uintptr_t)&ldsA[1][0];
  offB[0] = (unsigned int)(uintptr_t)&ldsB[0][0];
  offB[1] = (unsigned int)(uintptr_t)&ldsB[1][0];

  const int KSTEPS = K >> 5;
  if (wave == 0) {                                    // uniform: execz-skipped
    tdm_load_tile_2d(A  + (size_t)rblk * K + 0, offA[0], (unsigned)K);
    tdm_load_tile_2d(BT + (size_t)cblk * K + 0, offB[0], (unsigned)K);
    if (KSTEPS > 1) {
      tdm_load_tile_2d(A  + (size_t)rblk * K + 32, offA[1], (unsigned)K);
      tdm_load_tile_2d(BT + (size_t)cblk * K + 32, offB[1], (unsigned)K);
    }
  }
  for (int ks = 0; ks < KSTEPS; ks++) {
    int buf = ks & 1;
    if (wave == 0) {
      if (ks + 1 < KSTEPS) __builtin_amdgcn_s_wait_tensorcnt(2);
      else                 __builtin_amdgcn_s_wait_tensorcnt(0);
    }
    __syncthreads();                                   // tile visible to all
    v16bf af[4], bf_[4];
#pragma unroll
    for (int mt = 0; mt < 4; mt++) {
      int arow = (wave >> 1) * 64 + mt * 16 + lr;      // 0..127 in tile
      af[mt] = load_frag_a(&ldsA[buf][(size_t)arow * 32 + half * 8]);
    }
#pragma unroll
    for (int nt = 0; nt < 4; nt++) {
      int bcol = (wave & 1) * 64 + nt * 16 + lr;       // 0..127 in tile
      bf_[nt] = load_frag_b(&ldsB[buf][(size_t)bcol * 32 + half * 8]);
    }
#pragma unroll
    for (int mt = 0; mt < 4; mt++)
#pragma unroll
      for (int nt = 0; nt < 4; nt++)
        acc[mt][nt] = wmma_bf16(af[mt], bf_[nt], acc[mt][nt]);
    __syncthreads();                                   // done reading buf
    if (wave == 0 && ks + 2 < KSTEPS) {
      int kc = (ks + 2) * 32;
      tdm_load_tile_2d(A  + (size_t)rblk * K + kc, offA[buf], (unsigned)K);
      tdm_load_tile_2d(BT + (size_t)cblk * K + kc, offB[buf], (unsigned)K);
    }
  }
#else
  for (int kc = 0; kc < K; kc += 32) {
    v16bf af[4], bf_[4];
#pragma unroll
    for (int mt = 0; mt < 4; mt++)
      af[mt] = load_frag_a(A + (size_t)(rbase + mt * 16 + lr) * K + kc + half * 8);
#pragma unroll
    for (int nt = 0; nt < 4; nt++)
      bf_[nt] = load_frag_b(BT + (size_t)(cbase + nt * 16 + lr) * K + kc + half * 8);
#pragma unroll
    for (int mt = 0; mt < 4; mt++)
#pragma unroll
      for (int nt = 0; nt < 4; nt++)
        acc[mt][nt] = wmma_bf16(af[mt], bf_[nt], acc[mt][nt]);
  }
#endif

#pragma unroll
  for (int mt = 0; mt < 4; mt++)
#pragma unroll
    for (int nt = 0; nt < 4; nt++)
#pragma unroll
      for (int i = 0; i < 8; i++) {
        int row = rbase + mt * 16 + i + 8 * half;
        int col = cbase + nt * 16 + lr;
        float v = acc[mt][nt][i] * scale;
        if (WRITE_F32) {
          ((float*)C)[(size_t)row * N + col] = v;
        } else if (TRANS_C) {
          ((__bf16*)C)[(size_t)col * M + row] = (__bf16)v;
        } else {
          ((__bf16*)C)[(size_t)row * N + col] = (__bf16)v;
        }
      }
}

// ---------------------------------------------------------------------------
// Kernel 5: flash attention.  1 block per (b,h), 4 waves x 16 queries.
// ---------------------------------------------------------------------------
__global__ __launch_bounds__(128) void attn_kernel(
    const __bf16* __restrict__ q, const __bf16* __restrict__ Km,
    const __bf16* __restrict__ VT, const unsigned char* __restrict__ mask,
    __bf16* __restrict__ attnout) {
  int bh = blockIdx.x;
  int b = bh >> 3, h = bh & 7;
  int wave = threadIdx.x >> 5;
  int lane = threadIdx.x & 31;
  int half = lane >> 4;
  int lr   = lane & 15;

  __shared__ __align__(16) __bf16 lds_p[4][16][64];   // per-wave P tile

  const __bf16* qrow = q + (size_t)(b * N_LAT + wave * 16 + lr) * INNER + h * DHEAD;
  v16bf aq[2];
  aq[0] = load_frag_a(qrow + 0  + half * 8);
  aq[1] = load_frag_a(qrow + 32 + half * 8);

  float m[8], l[8];
  v8f oacc[4];
#pragma unroll
  for (int i = 0; i < 8; i++) { m[i] = NEG_BIG; l[i] = 0.f; }
#pragma unroll
  for (int t = 0; t < 4; t++) oacc[t] = vzero8();

  for (int kv0 = 0; kv0 < KVLEN; kv0 += 64) {
    // prefetch next K chunk (global_prefetch_b8) while computing this one
    if (kv0 + 64 < KVLEN) {
      int nk = kv0 + 64 + lane;
      __builtin_prefetch(Km + (size_t)(b * KVLEN + nk) * INNER + h * DHEAD, 0, 0);
      __builtin_prefetch(Km + (size_t)(b * KVLEN + nk + 32) * INNER + h * DHEAD, 0, 0);
    }
    // ---- S = (q*SCALE) @ K^T for 16 queries x 64 keys ----
    v8f s[4];
#pragma unroll
    for (int t = 0; t < 4; t++) s[t] = vzero8();
#pragma unroll
    for (int kc = 0; kc < 64; kc += 32) {
#pragma unroll
      for (int t = 0; t < 4; t++) {
        int key = kv0 + t * 16 + lr;    // lane = key column in C layout
        const __bf16* kp =
            Km + (size_t)(b * KVLEN + key) * INNER + h * DHEAD + kc + half * 8;
        v16bf bk = load_frag_b(kp);
        s[t] = wmma_bf16(aq[kc >> 5], bk, s[t]);
      }
    }
    // ---- mask + per-row max ----
    float cand[8];
#pragma unroll
    for (int i = 0; i < 8; i++) cand[i] = NEG_BIG;
#pragma unroll
    for (int t = 0; t < 4; t++) {
      int key = kv0 + t * 16 + lr;
      bool valid = (key >= F_FEAT) || (mask[(size_t)b * F_FEAT + key] == 0);
#pragma unroll
      for (int i = 0; i < 8; i++) {
        float v = valid ? s[t][i] : NEG_BIG;
        s[t][i] = v;
        cand[i] = fmaxf(cand[i], v);
      }
    }
#pragma unroll
    for (int off = 1; off < 16; off <<= 1)
#pragma unroll
      for (int i = 0; i < 8; i++)
        cand[i] = fmaxf(cand[i], __shfl_xor(cand[i], off, 32));
    // ---- online softmax update ----
    float sc[8], ps[8];
#pragma unroll
    for (int i = 0; i < 8; i++) {
      float mn = fmaxf(m[i], cand[i]);
      sc[i] = __expf(m[i] - mn);
      m[i] = mn;
      ps[i] = 0.f;
    }
#pragma unroll
    for (int t = 0; t < 4; t++)
#pragma unroll
      for (int i = 0; i < 8; i++) {
        float p = __expf(s[t][i] - m[i]);
        s[t][i] = p;
        ps[i] += p;
      }
#pragma unroll
    for (int off = 1; off < 16; off <<= 1)
#pragma unroll
      for (int i = 0; i < 8; i++) ps[i] += __shfl_xor(ps[i], off, 32);
#pragma unroll
    for (int i = 0; i < 8; i++) l[i] = l[i] * sc[i] + ps[i];
#pragma unroll
    for (int t = 0; t < 4; t++)
#pragma unroll
      for (int i = 0; i < 8; i++) oacc[t][i] *= sc[i];
    // ---- P (C layout) -> LDS -> A-layout bf16 fragments (same-wave order) --
#pragma unroll
    for (int t = 0; t < 4; t++)
#pragma unroll
      for (int i = 0; i < 8; i++)
        lds_p[wave][i + 8 * half][t * 16 + lr] = (__bf16)s[t][i];
    // ---- O += P @ V  (V^T rows contiguous over kv position) ----
#pragma unroll
    for (int kc = 0; kc < 64; kc += 32) {
      v16bf ap = load_frag_a(&lds_p[wave][lr][kc + half * 8]);
#pragma unroll
      for (int t = 0; t < 4; t++) {
        const __bf16* vp = VT +
            (size_t)(h * DHEAD + t * 16 + lr) * ((size_t)B_BATCH * KVLEN) +
            (size_t)b * KVLEN + kv0 + kc + half * 8;
        v16bf bv = load_frag_b(vp);
        oacc[t] = wmma_bf16(ap, bv, oacc[t]);
      }
    }
  }
#pragma unroll
  for (int t = 0; t < 4; t++)
#pragma unroll
    for (int i = 0; i < 8; i++) {
      int row = wave * 16 + i + 8 * half;
      float v = oacc[t][i] / l[i];
      attnout[(size_t)(b * N_LAT + row) * INNER + h * DHEAD + t * 16 + lr] =
          (__bf16)v;
    }
}

// ---------------------------------------------------------------------------
// Host launcher
// ---------------------------------------------------------------------------
extern "C" void kernel_launch(void* const* d_in, const int* in_sizes, int n_in,
                              void* d_out, int out_size, void* d_ws, size_t ws_size,
                              hipStream_t stream) {
  const float* x          = (const float*)d_in[0];
  const float* latents    = (const float*)d_in[1];
  const unsigned char* km = (const unsigned char*)d_in[2];  // bool mask (b,f)
  const float* kv_gate    = (const float*)d_in[3];
  const float* gamma_m    = (const float*)d_in[4];
  const float* beta_m     = (const float*)d_in[5];
  const float* gamma_l    = (const float*)d_in[6];
  const float* beta_l     = (const float*)d_in[7];
  const float* Wq         = (const float*)d_in[8];
  const float* Wk         = (const float*)d_in[9];
  const float* Wv         = (const float*)d_in[10];
  const float* Wout       = (const float*)d_in[11];
  float* out = (float*)d_out;

  const size_t M_KV = (size_t)B_BATCH * KVLEN;   // 66048
  char* ws = (char*)d_ws;
  size_t off = 0;
  auto take = [&](size_t bytes) -> char* {
    char* p = ws + off;
    off += (bytes + 255) & ~(size_t)255;
    return p;
  };
  __bf16* kv_bf   = (__bf16*)take(M_KV * D_MODEL * 2);                 // 135 MB
  __bf16* lat_bf  = (__bf16*)take((size_t)B_BATCH * N_LAT * D_MODEL * 2);
  __bf16* WqT     = (__bf16*)take((size_t)D_MODEL * INNER * 2);
  __bf16* WkT     = (__bf16*)take((size_t)D_MODEL * INNER * 2);
  __bf16* WvT     = (__bf16*)take((size_t)D_MODEL * INNER * 2);
  __bf16* WoutT   = (__bf16*)take((size_t)INNER * D_MODEL * 2);
  __bf16* qs      = (__bf16*)take((size_t)B_BATCH * N_LAT * INNER * 2);
  __bf16* Kmat    = (__bf16*)take(M_KV * INNER * 2);                   // 67 MB
  __bf16* VT      = (__bf16*)take(M_KV * INNER * 2);                   // 67 MB
  __bf16* attnout = (__bf16*)take((size_t)B_BATCH * N_LAT * INNER * 2);
  (void)ws_size; (void)n_in; (void)in_sizes; (void)out_size;

  // 1) weights -> bf16, transposed
  int wblk = (int)(((size_t)D_MODEL * INNER + 255) / 256);
  cvt_transpose_kernel<<<wblk, 256, 0, stream>>>(Wq,   WqT,   D_MODEL, INNER);
  cvt_transpose_kernel<<<wblk, 256, 0, stream>>>(Wk,   WkT,   D_MODEL, INNER);
  cvt_transpose_kernel<<<wblk, 256, 0, stream>>>(Wv,   WvT,   D_MODEL, INNER);
  cvt_transpose_kernel<<<wblk, 256, 0, stream>>>(Wout, WoutT, INNER, D_MODEL);

  // 2) LayerNorms + bf16 down-convert (single f32 pass over x)
  ln_media_kernel<<<B_BATCH * F_FEAT, 256, 0, stream>>>(x, gamma_m, beta_m, kv_bf);
  ln_latents_kernel<<<B_BATCH * N_LAT, 256, 0, stream>>>(
      latents, gamma_l, beta_l, kv_gate, lat_bf, kv_bf);

  // 3) projections (bf16 WMMA + TDM staging). Q has SCALE folded; V transposed.
  gemm_bf16_kernel<false, false><<<dim3(4, 4), 128, 0, stream>>>(
      lat_bf, WqT, qs, B_BATCH * N_LAT, INNER, D_MODEL, SCALE_Q);
  gemm_bf16_kernel<false, false><<<dim3((int)(M_KV / 128), 4), 128, 0, stream>>>(
      kv_bf, WkT, Kmat, (int)M_KV, INNER, D_MODEL, 1.0f);
  gemm_bf16_kernel<false, true><<<dim3((int)(M_KV / 128), 4), 128, 0, stream>>>(
      kv_bf, WvT, VT, (int)M_KV, INNER, D_MODEL, 1.0f);

  // 4) flash attention, one block per (b, h)
  attn_kernel<<<B_BATCH * HEADS, 128, 0, stream>>>(qs, Kmat, VT, km, attnout);

  // 5) output projection -> f32
  gemm_bf16_kernel<true, false><<<dim3(4, 8), 128, 0, stream>>>(
      attnout, WoutT, out, B_BATCH * N_LAT, D_MODEL, INNER, 1.0f);
}
